// PTSeg_Balance_Prior_67714454389202
// MI455X (gfx1250) — matrix operand order repaired and checked by reading, not compiled, not run on hardware
//
#include <hip/hip_runtime.h>

// ---------------------------------------------------------------------------
// Point Transformer segment encoder for MI455X (gfx1250, wave32, WMMA + TDM).
// - All channel-mixing GEMMs: v_wmma_f32_16x16x32_f16, f32 accumulate,
//   fused bias/BN/residual/ReLU epilogues.
// - Weights are pre-swizzled once into f16 WMMA-fragment tile order; each
//   32x32 B tile is a contiguous 2KB block fetched by the Tensor Data Mover
//   (tensor_load_to_lds + s_wait_tensorcnt) while the current WMMA executes.
// - Activations tiles (A) staged f32->f16 through LDS, double buffered,
//   one barrier per K step, global_prefetch for the next K tile.
// ---------------------------------------------------------------------------

typedef __attribute__((ext_vector_type(16))) _Float16 v16h;
typedef __attribute__((ext_vector_type(8)))  float    v8f;
typedef __attribute__((ext_vector_type(4)))  unsigned int v4u;
typedef __attribute__((ext_vector_type(8)))  int v8i;
typedef __attribute__((ext_vector_type(4)))  int v4i;

#define EPSBN 1e-5f
#define NCLS 13
#define USE_TDM 1

// ---- A tile loader: global f32 -> LDS f16, pre-swizzled to fragment order --
// A tile: 64x32. element (r,c): lane=(r&15)|((c>>3&1)<<4), slot=(c&7)|((c>>4)<<3)
__device__ __forceinline__ void load_a(
    const float* __restrict__ X, int N, int K, int n0, int k0, int tid,
    _Float16* __restrict__ Ah)
{
  const bool fastA = (n0 + 64 <= N) && (k0 + 32 <= K) && ((K & 3) == 0);
  if (fastA) {
    const int r = tid >> 2, quad = tid & 3;
    const float* src = X + (size_t)(n0 + r) * K + k0 + quad * 8;
    float4 f0 = ((const float4*)src)[0];
    float4 f1 = ((const float4*)src)[1];
    if (quad == 0 && k0 + 64 <= K) __builtin_prefetch(src + 32, 0, 1);
    float vals[8] = {f0.x, f0.y, f0.z, f0.w, f1.x, f1.y, f1.z, f1.w};
#pragma unroll
    for (int u = 0; u < 8; ++u) {
      int c = quad * 8 + u;
      int lane = (r & 15) | (((c >> 3) & 1) << 4);
      int slot = (c & 7) | ((c >> 4) << 3);
      Ah[(((r >> 4) * 32) + lane) * 16 + slot] = (_Float16)vals[u];
    }
  } else {
    for (int t = tid; t < 64 * 32; t += 256) {
      int r = t >> 5, c = t & 31;
      int gn = n0 + r, gk = k0 + c;
      float v = (gn < N && gk < K) ? X[(size_t)gn * K + gk] : 0.f;
      int lane = (r & 15) | (((c >> 3) & 1) << 4);
      int slot = (c & 7) | ((c >> 4) << 3);
      Ah[(((r >> 4) * 32) + lane) * 16 + slot] = (_Float16)v;
    }
  }
}

// ---- B tile: contiguous 2KB (pre-swizzled f16) -> LDS via TDM -------------
__device__ __forceinline__ void copy_b_tile(const _Float16* __restrict__ src,
                                            void* lds_dst, int tid)
{
#if USE_TDM
  if (tid < 32) {   // one TDM op issued by wave 0 (EXEC ignored by TDM)
    unsigned long long ga = (unsigned long long)(size_t)src;
    unsigned int lo = (unsigned int)(size_t)lds_dst;
    v4u g0;
    g0[0] = 1u;                                     // count=1, user descriptor
    g0[1] = lo;                                     // lds_addr
    g0[2] = (unsigned int)ga;                       // global_addr[31:0]
    g0[3] = (unsigned int)((ga >> 32) & 0x01FFFFFFu) | (2u << 30); // addr hi, type=2
    v8i g1;
    g1[0] = (int)(2u << 16);                        // data_size = 4B
    g1[1] = (int)((512u & 0xFFFFu) << 16);          // tensor_dim0[15:0] = 512
    g1[2] = (int)((512u >> 16) | (1u << 16));       // tensor_dim0 hi | tensor_dim1=1
    g1[3] = (int)(512u << 16);                      // tile_dim0 = 512
    g1[4] = 0;                                      // tile_dim1/2 unused
    g1[5] = 512;                                    // tensor_dim0_stride
    g1[6] = 0; g1[7] = 0;
    v4i z4 = {0, 0, 0, 0};
    v8i z8 = {0, 0, 0, 0, 0, 0, 0, 0};
    __builtin_amdgcn_tensor_load_to_lds(g0, g1, z4, z4, z8, 0);
  }
#else
  ((uint2*)lds_dst)[tid] = ((const uint2*)src)[tid];
#endif
}

// ---- weight pre-swizzle: W (KxM f32) -> per 32x32 tile fragment-order f16 --
// layout: tile (kt, mt) at ((mt*nkt)+kt)*1024 halves; within tile:
// element (k,m): ((m>>4)*32 + ((m&15)|(((k>>3)&1)<<4)))*16 + ((k&7)|((k>>4)<<3))
__global__ void swizzle_w(const float* __restrict__ W, _Float16* __restrict__ O,
                          int K, int M)
{
  const int kt = blockIdx.x, mt = blockIdx.y, nkt = gridDim.x;
  _Float16* dst = O + ((size_t)mt * nkt + kt) * 1024;
  for (int t = threadIdx.x; t < 1024; t += 256) {
    int k = t >> 5, m = t & 31;
    int gk = kt * 32 + k, gm = mt * 32 + m;
    float v = (gk < K && gm < M) ? W[(size_t)gk * M + gm] : 0.f;
    int lane = (m & 15) | (((k >> 3) & 1) << 4);
    int slot = (k & 7) | ((k >> 4) << 3);
    dst[(((m >> 4) * 32) + lane) * 16 + slot] = (_Float16)v;
  }
}

// ------------------------------ WMMA GEMM ----------------------------------
// Y[N,M] = act( BN( X[N,K] @ W[K,M] + bias ) + R ), W pre-swizzled f16.
__global__ __launch_bounds__(256) void gemm_bn_act(
    const float* __restrict__ X, const _Float16* __restrict__ Wsw,
    const float* __restrict__ bias,
    const float* __restrict__ bn_g, const float* __restrict__ bn_b,
    const float* __restrict__ bn_rm, const float* __restrict__ bn_rv,
    const float* __restrict__ R, float* __restrict__ Y,
    int N, int K, int M, int do_relu)
{
  __shared__ v16h Av[2][128];  // [buf][wrow(4) * 32 lanes]
  __shared__ v16h Bv[2][64];   // [buf][wcol(2) * 32 lanes]
  const int tid  = threadIdx.x;
  const int lane = tid & 31;
  const int wave = tid >> 5;
  const int wrow = wave >> 1;   // 0..3
  const int wcol = wave & 1;    // 0..1
  const int n0 = blockIdx.x * 64;
  const int mt = blockIdx.y;
  const int m0 = mt * 32;

  const int nk = (K + 31) >> 5;
  load_a(X, N, K, n0, 0, tid, (_Float16*)Av[0]);
  copy_b_tile(Wsw + (size_t)mt * nk * 1024, (_Float16*)Bv[0], tid);
  __builtin_amdgcn_s_wait_tensorcnt(0);
  __syncthreads();

  v8f acc = {};
  int cur = 0;
  for (int it = 0; it < nk; ++it) {
    if (it + 1 < nk) {
      load_a(X, N, K, n0, (it + 1) << 5, tid, (_Float16*)Av[1 - cur]);
      copy_b_tile(Wsw + ((size_t)mt * nk + it + 1) * 1024, (_Float16*)Bv[1 - cur], tid);
    }
    v16h af = Av[cur][wrow * 32 + lane];
    v16h bf = Bv[cur][wcol * 32 + lane];
    acc = __builtin_amdgcn_wmma_f32_16x16x32_f16(
        false, af, false, bf, (short)0, acc, false, false);
    __builtin_amdgcn_s_wait_tensorcnt(0);
    __syncthreads();
    cur ^= 1;
  }

  // C/D layout: VGPR r -> row (wrow*16 + r + 8*(lane>=16)), col = wcol*16+lane%16
  const int colD  = m0 + wcol * 16 + (lane & 15);
  const int rbase = n0 + wrow * 16 + ((lane >> 4) ? 8 : 0);
  float bv = 0.f, scale = 1.f, shift = 0.f;
  const bool has_bn = (bn_g != nullptr);
  if (colD < M) {
    if (bias) bv = bias[colD];
    if (has_bn) {
      scale = bn_g[colD] * rsqrtf(bn_rv[colD] + EPSBN);
      shift = bn_b[colD] - bn_rm[colD] * scale;
    }
  }
#pragma unroll
  for (int r = 0; r < 8; ++r) {
    int rowD = rbase + r;
    if (rowD < N && colD < M) {
      float v = acc[r] + bv;
      if (has_bn) v = v * scale + shift;
      if (R) v += R[(size_t)rowD * M + colD];
      if (do_relu) v = fmaxf(v, 0.f);
      Y[(size_t)rowD * M + colD] = v;
    }
  }
}

// ------------------------------ KNN ----------------------------------------
template <int NS>
__global__ __launch_bounds__(256) void knn_kernel(
    const float* __restrict__ qp, const float* __restrict__ cp,
    int nq, int nc, int* __restrict__ out)
{
  const int seg = blockIdx.x;
  const int q = blockIdx.y * 256 + threadIdx.x;
  const float* qs = qp + (size_t)seg * nq * 3;
  const float* cs = cp + (size_t)seg * nc * 3;
  const bool active = q < nq;
  float qx = 0, qy = 0, qz = 0;
  if (active) { qx = qs[q * 3]; qy = qs[q * 3 + 1]; qz = qs[q * 3 + 2]; }
  float bd[NS];
  int bi[NS];
#pragma unroll
  for (int t = 0; t < NS; ++t) { bd[t] = 1e30f; bi[t] = 0; }
  __shared__ float tx[256], ty[256], tz[256];
  for (int base = 0; base < nc; base += 256) {
    int c = base + threadIdx.x;
    if (c < nc) { tx[threadIdx.x] = cs[c * 3]; ty[threadIdx.x] = cs[c * 3 + 1]; tz[threadIdx.x] = cs[c * 3 + 2]; }
    __syncthreads();
    int lim = nc - base; if (lim > 256) lim = 256;
    if (active) {
      for (int t = 0; t < lim; ++t) {
        float dx = qx - tx[t], dy = qy - ty[t], dz = qz - tz[t];
        float d = dx * dx + dy * dy + dz * dz;
        if (d < bd[NS - 1]) {
          bd[NS - 1] = d; bi[NS - 1] = base + t;
#pragma unroll
          for (int u = NS - 1; u > 0; --u) {
            if (bd[u] < bd[u - 1]) {
              float td_ = bd[u]; bd[u] = bd[u - 1]; bd[u - 1] = td_;
              int   ti_ = bi[u]; bi[u] = bi[u - 1]; bi[u - 1] = ti_;
            }
          }
        }
      }
    }
    __syncthreads();
  }
  if (active) {
#pragma unroll
    for (int t = 0; t < NS; ++t)
      out[((size_t)seg * nq + q) * NS + t] = bi[t];
  }
}

// ------------------------------ FPS ----------------------------------------
__global__ __launch_bounds__(256) void fps_kernel(
    const float* __restrict__ p, int n, int m, int* __restrict__ out)
{
  const int seg = blockIdx.x, tid = threadIdx.x;
  const float* ps = p + (size_t)seg * n * 3;
  float px[16], py[16], pz[16], pd[16];
  const int chunks = (n + 255) >> 8;
  for (int u = 0; u < chunks; ++u) {
    int i = tid + (u << 8);
    if (i < n) { px[u] = ps[i * 3]; py[u] = ps[i * 3 + 1]; pz[u] = ps[i * 3 + 2]; pd[u] = 1e10f; }
    else { px[u] = 0; py[u] = 0; pz[u] = 0; pd[u] = -1e30f; }
  }
  __shared__ float rmax[256];
  __shared__ int   rarg[256];
  __shared__ float lxyz[3];
  if (tid == 0) out[(size_t)seg * m] = 0;
  int last = 0;
  for (int step = 1; step < m; ++step) {
    if ((last & 255) == tid) { int u = last >> 8; lxyz[0] = px[u]; lxyz[1] = py[u]; lxyz[2] = pz[u]; }
    __syncthreads();
    const float lx = lxyz[0], ly = lxyz[1], lz = lxyz[2];
    float bmax = -1e30f; int barg = 0;
    for (int u = 0; u < chunks; ++u) {
      float dx = px[u] - lx, dy = py[u] - ly, dz = pz[u] - lz;
      float d = dx * dx + dy * dy + dz * dz;
      float nd = fminf(pd[u], d);
      pd[u] = nd;
      if (nd > bmax) { bmax = nd; barg = tid + (u << 8); }
    }
    rmax[tid] = bmax; rarg[tid] = barg;
    __syncthreads();
    for (int off = 128; off; off >>= 1) {
      if (tid < off) {
        if (rmax[tid + off] > rmax[tid] ||
            (rmax[tid + off] == rmax[tid] && rarg[tid + off] < rarg[tid])) {
          rmax[tid] = rmax[tid + off]; rarg[tid] = rarg[tid + off];
        }
      }
      __syncthreads();
    }
    last = rarg[0];
    if (tid == 0) out[(size_t)seg * m + step] = last;
    __syncthreads();
  }
}

// ------------------------- TD helpers --------------------------------------
__global__ void gather_newp(const float* __restrict__ p, const int* __restrict__ fi,
                            int n, int m, float* __restrict__ pn)
{
  int t = blockIdx.x * blockDim.x + threadIdx.x;
  if (t >= NCLS * m) return;
  int seg = t / m;
  int j = fi[t];
  const float* src = p + ((size_t)seg * n + j) * 3;
  float* dst = pn + (size_t)t * 3;
  dst[0] = src[0]; dst[1] = src[1]; dst[2] = src[2];
}

__global__ void build_g(const float* __restrict__ p, const float* __restrict__ x,
                        const float* __restrict__ newp, const int* __restrict__ idx,
                        int n, int m, int ns, int cin, float* __restrict__ G)
{
  long row = blockIdx.x;                    // over NCLS*m*ns
  int seg = (int)(row / ((long)m * ns));
  long rem = row % ((long)m * ns);
  int mi = (int)(rem / ns);
  int j = idx[row];
  const float* pj = p + ((size_t)seg * n + j) * 3;
  const float* np = newp + ((size_t)seg * m + mi) * 3;
  float* g = G + (size_t)row * (3 + cin);
  if (threadIdx.x < 3) g[threadIdx.x] = pj[threadIdx.x] - np[threadIdx.x];
  for (int c = threadIdx.x; c < cin; c += blockDim.x)
    g[3 + c] = x[((size_t)seg * n + j) * cin + c];
}

__global__ void max_ns(const float* __restrict__ H, long rows, int ns, int C,
                       float* __restrict__ out)
{
  long t = (long)blockIdx.x * blockDim.x + threadIdx.x;
  if (t >= rows * C) return;
  long r = t / C; int c = (int)(t % C);
  float v = -1e30f;
  for (int nn = 0; nn < ns; ++nn)
    v = fmaxf(v, H[((size_t)r * ns + nn) * C + c]);
  out[(size_t)r * C + c] = v;
}

__global__ void bn_relu(const float* __restrict__ X, const float* __restrict__ g,
                        const float* __restrict__ b, const float* __restrict__ rm,
                        const float* __restrict__ rv, float* __restrict__ Y,
                        long total, int C)
{
  long t = (long)blockIdx.x * blockDim.x + threadIdx.x;
  if (t >= total) return;
  int c = (int)(t % C);
  float s = g[c] * rsqrtf(rv[c] + EPSBN);
  Y[t] = fmaxf((X[t] - rm[c]) * s + b[c], 0.f);
}

// --------------------------- Fused vector attention ------------------------
template <int C, int NS>
__global__ __launch_bounds__(256) void attn_kernel(
    const float* __restrict__ p, const float* __restrict__ q,
    const float* __restrict__ k, const float* __restrict__ v,
    const int* __restrict__ idx, int nseg,
    const float* __restrict__ Wp1, const float* __restrict__ bp1,
    const float* __restrict__ pbn_g, const float* __restrict__ pbn_b,
    const float* __restrict__ pbn_rm, const float* __restrict__ pbn_rv,
    const float* __restrict__ Wp2, const float* __restrict__ bp2,
    const float* __restrict__ w1g, const float* __restrict__ w1b,
    const float* __restrict__ w1rm, const float* __restrict__ w1rv,
    const float* __restrict__ Wl1, const float* __restrict__ bl1,
    const float* __restrict__ w2g, const float* __restrict__ w2b,
    const float* __restrict__ w2rm, const float* __restrict__ w2rv,
    const float* __restrict__ Wl2, const float* __restrict__ bl2,
    float* __restrict__ y)
{
  constexpr int C8 = C / 8;
  __shared__ float q_s[C];
  __shared__ float t1_s[NS * C];
  __shared__ float t2_s[NS * C8];
  __shared__ float wl_s[NS * C8];
  __shared__ float h_s[NS * 3];
  const int i = blockIdx.x;
  const int seg = i / nseg;
  const int tid = threadIdx.x;
  const float pix = p[(size_t)i * 3], piy = p[(size_t)i * 3 + 1], piz = p[(size_t)i * 3 + 2];
  int jloc[NS];
#pragma unroll
  for (int nn = 0; nn < NS; ++nn) jloc[nn] = idx[(size_t)i * NS + nn];
  for (int c = tid; c < C; c += 256) q_s[c] = q[(size_t)i * C + c];
  __syncthreads();
  for (int nn = 0; nn < NS; ++nn) {
    size_t j = (size_t)seg * nseg + jloc[nn];
    float prx = p[j * 3] - pix, pry = p[j * 3 + 1] - piy, prz = p[j * 3 + 2] - piz;
    float h[3];
#pragma unroll
    for (int t = 0; t < 3; ++t) {
      float a = prx * Wp1[0 * 3 + t] + pry * Wp1[1 * 3 + t] + prz * Wp1[2 * 3 + t] + bp1[t];
      float s = pbn_g[t] * rsqrtf(pbn_rv[t] + EPSBN);
      h[t] = fmaxf((a - pbn_rm[t]) * s + pbn_b[t], 0.f);
    }
    if (tid == 0) { h_s[nn * 3] = h[0]; h_s[nn * 3 + 1] = h[1]; h_s[nn * 3 + 2] = h[2]; }
    for (int c = tid; c < C; c += 256) {
      float pe = h[0] * Wp2[c] + h[1] * Wp2[C + c] + h[2] * Wp2[2 * C + c] + bp2[c];
      float w = k[j * C + c] - q_s[c] + pe;
      float s = w1g[c] * rsqrtf(w1rv[c] + EPSBN);
      t1_s[nn * C + c] = fmaxf((w - w1rm[c]) * s + w1b[c], 0.f);
    }
  }
  __syncthreads();
  for (int o = tid; o < NS * C8; o += 256) {
    int nn = o / C8, jo = o % C8;
    float acc = bl1[jo];
    for (int c = 0; c < C; ++c) acc += t1_s[nn * C + c] * Wl1[(size_t)c * C8 + jo];
    float s = w2g[jo] * rsqrtf(w2rv[jo] + EPSBN);
    t2_s[o] = fmaxf((acc - w2rm[jo]) * s + w2b[jo], 0.f);
  }
  __syncthreads();
  for (int o = tid; o < NS * C8; o += 256) {
    int nn = o / C8, jo = o % C8;
    float acc = bl2[jo];
#pragma unroll 4
    for (int c = 0; c < C8; ++c) acc += t2_s[nn * C8 + c] * Wl2[c * C8 + jo];
    wl_s[o] = acc;
  }
  __syncthreads();
  for (int jo = tid; jo < C8; jo += 256) {
    float mx = -1e30f;
#pragma unroll
    for (int nn = 0; nn < NS; ++nn) mx = fmaxf(mx, wl_s[nn * C8 + jo]);
    float e[NS], ssum = 0.f;
#pragma unroll
    for (int nn = 0; nn < NS; ++nn) { e[nn] = expf(wl_s[nn * C8 + jo] - mx); ssum += e[nn]; }
    float inv = 1.f / ssum;
#pragma unroll
    for (int nn = 0; nn < NS; ++nn) wl_s[nn * C8 + jo] = e[nn] * inv;
  }
  __syncthreads();
  for (int c = tid; c < C; c += 256) {
    int jo = c & (C8 - 1);
    float acc = 0.f;
#pragma unroll
    for (int nn = 0; nn < NS; ++nn) {
      size_t j = (size_t)seg * nseg + jloc[nn];
      float pe = h_s[nn * 3] * Wp2[c] + h_s[nn * 3 + 1] * Wp2[C + c] +
                 h_s[nn * 3 + 2] * Wp2[2 * C + c] + bp2[c];
      acc += (v[j * C + c] + pe) * wl_s[nn * C8 + jo];
    }
    y[(size_t)i * C + c] = acc;
  }
}

// ----------------------------- head kernels --------------------------------
__global__ void l2norm_rows(const float* __restrict__ X, float* __restrict__ Y, int C)
{
  __shared__ float red[64];
  const int r = blockIdx.x, tid = threadIdx.x;
  float s = 0.f;
  for (int c = tid; c < C; c += 64) { float v = X[(size_t)r * C + c]; s += v * v; }
  red[tid] = s; __syncthreads();
  for (int off = 32; off; off >>= 1) { if (tid < off) red[tid] += red[tid + off]; __syncthreads(); }
  float nrm = fmaxf(sqrtf(red[0]), 1e-12f);
  for (int c = tid; c < C; c += 64) Y[(size_t)r * C + c] = X[(size_t)r * C + c] / nrm;
}

__global__ void write_cur(const float* __restrict__ feat, float* __restrict__ out)
{
  const int r = blockIdx.x, c = threadIdx.x;
  if (c < 32) out[r * 33 + c] = feat[r * 32 + c];
  else if (c == 32) out[r * 33 + 32] = (float)(r / 16);
}

__global__ void new_prior_kernel(const float* __restrict__ feat,
                                 const float* __restrict__ ema,
                                 float* __restrict__ out)
{
  const int cls = blockIdx.x, c = threadIdx.x; // 32 threads
  float mean = 0.f;
  for (int r = 0; r < 16; ++r) mean += feat[(cls * 16 + r) * 32 + c];
  mean *= (1.f / 16.f);
  float u = 0.999f * ema[cls * 32 + c] + 0.001f * mean;
  __shared__ float red[32];
  red[c] = u * u; __syncthreads();
  for (int off = 16; off; off >>= 1) { if (c < off) red[c] += red[c + off]; __syncthreads(); }
  float nrm = fmaxf(sqrtf(red[0]), 1e-12f);
  out[cls * 32 + c] = u / nrm;
}

// ----------------------------- host side -----------------------------------
struct BNP  { const float *b, *g, *rm, *rv; };
struct LinP { const float *b, *w; };
struct BlockP {
  BNP bn1, bn2, bn3;
  LinP k, lin1, lin3, p1, p2;
  BNP pbn;
  LinP q, v;
  BNP wbn1, wbn2;
  LinP wl1, wl2;
};

static void gemm(hipStream_t s, const float* X, const _Float16* Wsw, const float* bias,
                 const BNP* bn, const float* R, float* Y, int N, int K, int M, int relu)
{
  dim3 g((N + 63) / 64, (M + 31) / 32);
  gemm_bn_act<<<g, 256, 0, s>>>(X, Wsw, bias,
                                bn ? bn->g : nullptr, bn ? bn->b : nullptr,
                                bn ? bn->rm : nullptr, bn ? bn->rv : nullptr,
                                R, Y, N, K, M, relu);
}

extern "C" void kernel_launch(void* const* d_in, const int* in_sizes, int n_in,
                              void* d_out, int out_size, void* d_ws, size_t ws_size,
                              hipStream_t stream)
{
  (void)in_sizes; (void)n_in; (void)out_size; (void)ws_size;
  int li = 0;
  const int* labels = (const int*)d_in[li++]; (void)labels; // sorted, equal counts -> identity regroup
  const float* p0 = (const float*)d_in[li++];

  auto getf = [&]() { return (const float*)d_in[li++]; };
  auto getBN = [&](BNP& o) { o.b = getf(); o.g = getf(); o.rm = getf(); o.rv = getf(); };
  auto getLinB = [&](LinP& o) { o.b = getf(); o.w = getf(); };
  auto getLinNB = [&](LinP& o) { o.b = nullptr; o.w = getf(); };

  // params: dict keys sorted -> "proj" before "stages"
  BNP proj_bn[3]; LinP proj_lin[3];
  for (int i = 0; i < 3; ++i) { getBN(proj_bn[i]); getLinB(proj_lin[i]); }

  const int NBm1[5] = {1, 2, 3, 5, 2};
  BlockP blocks[13]; BNP td_bn[5]; const float* td_w[5];
  int blk_off[5]; int bo = 0;
  for (int s = 0; s < 5; ++s) {
    blk_off[s] = bo;
    for (int b = 0; b < NBm1[s]; ++b) {
      BlockP& B = blocks[bo++];
      getBN(B.bn1); getBN(B.bn2); getBN(B.bn3);
      getLinB(B.k); getLinNB(B.lin1); getLinNB(B.lin3);
      getLinB(B.p1); getLinB(B.p2); getBN(B.pbn);
      getLinB(B.q); getLinB(B.v);
      getBN(B.wbn1); getBN(B.wbn2);
      getLinB(B.wl1); getLinB(B.wl2);
    }
    getBN(td_bn[s]); td_w[s] = getf();
  }
  const float* prior_ema = getf();
  const float* x0 = getf();

  // workspace carve
  char* wsp = (char*)d_ws; size_t off = 0;
  auto alloc = [&](size_t count) -> float* {   // count in floats (4B units)
    float* r = (float*)(wsp + off);
    off += ((count * 4) + 255) & ~(size_t)255;
    return r;
  };
  const size_t MAXNC = (size_t)53248 * 32;
  float* fXa = alloc(MAXNC); float* fXb = alloc(MAXNC);
  float* fX1 = alloc(MAXNC); float* fQ = alloc(MAXNC);
  float* fK  = alloc(MAXNC); float* fV = alloc(MAXNC);
  float* fY  = alloc(MAXNC);
  float* pb0 = alloc((size_t)NCLS * 4096 * 3);
  float* pb1 = alloc((size_t)NCLS * 4096 * 3);
  float* G   = alloc(7454720);    // 13*1024*16*35
  float* H   = alloc(13631488);   // 13*1024*16*64
  int* knnIdx = (int*)alloc((size_t)53248 * 16);
  int* fpsIdx = (int*)alloc((size_t)NCLS * 1024);

  // weight pre-swizzle (f32 KxM -> f16 tile-fragment order, K/M padded to 32)
  auto prep_w = [&](const float* w, int K, int M) -> const _Float16* {
    int nkt = (K + 31) >> 5, nmt = (M + 31) >> 5;
    size_t halves = (size_t)nkt * nmt * 1024;
    _Float16* o = (_Float16*)alloc((halves + 1) / 2);
    swizzle_w<<<dim3(nkt, nmt), 256, 0, stream>>>(w, o, K, M);
    return o;
  };
  const int WID[5] = {32, 64, 128, 256, 512};
  const int NSs[5] = {8, 16, 16, 16, 16};
  const _Float16* td_wsw[5];
  td_wsw[0] = prep_w(td_w[0], 6, 32);
  td_wsw[1] = prep_w(td_w[1], 35, 64);
  td_wsw[2] = prep_w(td_w[2], 67, 128);
  td_wsw[3] = prep_w(td_w[3], 131, 256);
  td_wsw[4] = prep_w(td_w[4], 259, 512);
  const _Float16* blk_wsw[13][5];  // lin1, q, k, v, lin3
  bo = 0;
  for (int s = 0; s < 5; ++s) {
    const int C = WID[s];
    for (int b = 0; b < NBm1[s]; ++b) {
      const BlockP& B = blocks[bo];
      blk_wsw[bo][0] = prep_w(B.lin1.w, C, C);
      blk_wsw[bo][1] = prep_w(B.q.w, C, C);
      blk_wsw[bo][2] = prep_w(B.k.w, C, C);
      blk_wsw[bo][3] = prep_w(B.v.w, C, C);
      blk_wsw[bo][4] = prep_w(B.lin3.w, C, C);
      ++bo;
    }
  }
  const _Float16* proj_wsw[3];
  proj_wsw[0] = prep_w(proj_lin[0].w, 512, 256);
  proj_wsw[1] = prep_w(proj_lin[1].w, 256, 128);
  proj_wsw[2] = prep_w(proj_lin[2].w, 128, 32);

  int nseg = 4096;
  const float* pcur = p0;
  float* xping[2] = {fXa, fXb}; int xi = 1;
  float* ppick[2] = {pb0, pb1}; int pi = 0;

  // stage 0 TransitionDown (stride 1): x = relu(bn(x0 @ W))
  gemm(stream, x0, td_wsw[0], nullptr, &td_bn[0], nullptr, xping[0],
       NCLS * 4096, 6, 32, 1);
  const float* xcur = xping[0];

  for (int s = 0; s < 5; ++s) {
    const int C = WID[s], NSS = NSs[s];
    if (s > 0) {
      const int n = nseg, m = nseg / 4, Cin = WID[s - 1];
      const int tot_m = NCLS * m;
      fps_kernel<<<NCLS, 256, 0, stream>>>(pcur, n, m, fpsIdx);
      gather_newp<<<(tot_m + 255) / 256, 256, 0, stream>>>(pcur, fpsIdx, n, m, ppick[pi]);
      float* pnew = ppick[pi]; pi ^= 1;
      knn_kernel<16><<<dim3(NCLS, (m + 255) / 256), 256, 0, stream>>>(pnew, pcur, m, n, knnIdx);
      const int rows = tot_m * 16;
      build_g<<<rows, 64, 0, stream>>>(pcur, xcur, pnew, knnIdx, n, m, 16, Cin, G);
      gemm(stream, G, td_wsw[s], nullptr, &td_bn[s], nullptr, H, rows, 3 + Cin, C, 1);
      max_ns<<<(int)(((long)tot_m * C + 255) / 256), 256, 0, stream>>>(H, tot_m, 16, C, xping[xi]);
      xcur = xping[xi]; xi ^= 1;
      pcur = pnew; nseg = m;
    }
    const int npts = NCLS * nseg;
    if (NSS == 8)
      knn_kernel<8><<<dim3(NCLS, (nseg + 255) / 256), 256, 0, stream>>>(pcur, pcur, nseg, nseg, knnIdx);
    else
      knn_kernel<16><<<dim3(NCLS, (nseg + 255) / 256), 256, 0, stream>>>(pcur, pcur, nseg, nseg, knnIdx);

    for (int b = 0; b < NBm1[s]; ++b) {
      const int gb = blk_off[s] + b;
      const BlockP& B = blocks[gb];
      gemm(stream, xcur, blk_wsw[gb][0], nullptr, &B.bn1, nullptr, fX1, npts, C, C, 1);
      gemm(stream, fX1, blk_wsw[gb][1], B.q.b, nullptr, nullptr, fQ, npts, C, C, 0);
      gemm(stream, fX1, blk_wsw[gb][2], B.k.b, nullptr, nullptr, fK, npts, C, C, 0);
      gemm(stream, fX1, blk_wsw[gb][3], B.v.b, nullptr, nullptr, fV, npts, C, C, 0);
#define ATTN_ARGS pcur, fQ, fK, fV, knnIdx, nseg, \
      B.p1.w, B.p1.b, B.pbn.g, B.pbn.b, B.pbn.rm, B.pbn.rv, B.p2.w, B.p2.b, \
      B.wbn1.g, B.wbn1.b, B.wbn1.rm, B.wbn1.rv, B.wl1.w, B.wl1.b, \
      B.wbn2.g, B.wbn2.b, B.wbn2.rm, B.wbn2.rv, B.wl2.w, B.wl2.b, fY
      switch (s) {
        case 0: attn_kernel<32, 8><<<npts, 256, 0, stream>>>(ATTN_ARGS); break;
        case 1: attn_kernel<64, 16><<<npts, 256, 0, stream>>>(ATTN_ARGS); break;
        case 2: attn_kernel<128, 16><<<npts, 256, 0, stream>>>(ATTN_ARGS); break;
        case 3: attn_kernel<256, 16><<<npts, 256, 0, stream>>>(ATTN_ARGS); break;
        case 4: attn_kernel<512, 16><<<npts, 256, 0, stream>>>(ATTN_ARGS); break;
      }
#undef ATTN_ARGS
      bn_relu<<<(int)(((long)npts * C + 255) / 256), 256, 0, stream>>>(
          fY, B.bn2.g, B.bn2.b, B.bn2.rm, B.bn2.rv, fX1, (long)npts * C, C);
      gemm(stream, fX1, blk_wsw[gb][4], nullptr, &B.bn3, xcur, xping[xi], npts, C, C, 1);
      xcur = xping[xi]; xi ^= 1;
    }
  }

  // projection head: 208x512 -> 256 -> 128 -> 32, each relu(bn(lin(.)))
  gemm(stream, xcur, proj_wsw[0], proj_lin[0].b, &proj_bn[0], nullptr, fQ, 208, 512, 256, 1);
  gemm(stream, fQ,   proj_wsw[1], proj_lin[1].b, &proj_bn[1], nullptr, fK, 208, 256, 128, 1);
  gemm(stream, fK,   proj_wsw[2], proj_lin[2].b, &proj_bn[2], nullptr, fV, 208, 128, 32, 1);
  l2norm_rows<<<208, 64, 0, stream>>>(fV, fY, 32);

  float* out = (float*)d_out;
  write_cur<<<208, 64, 0, stream>>>(fY, out);                    // current_prior (208 x 33)
  new_prior_kernel<<<13, 32, 0, stream>>>(fY, prior_ema, out + 208 * 33); // new_prior (13 x 32)
}
